// MambaBlock_10196252360834
// MI455X (gfx1250) — matrix operand order repaired and checked by reading, not compiled, not run on hardware
//
#include <hip/hip_runtime.h>
#include <math.h>

// Problem constants (match reference)
#define D_MODEL 1024
#define D_STATE 16
#define D_CONV  4
#define D_INNER 2048
#define DT_RANK 64
#define B_SZ    2
#define SEQ     2048
#define M_TOT   (B_SZ * SEQ)             // 4096 token rows
#define DBC_COLS (DT_RANK + 2 * D_STATE) // 96

typedef __attribute__((ext_vector_type(16))) __bf16 v16bf;
typedef __attribute__((ext_vector_type(8)))  float  v8f;

// fp32 -> (hi, lo) bf16 split, round-to-nearest-even in integer ops.
__device__ __forceinline__ void split_bf16(float x, __bf16& hi, __bf16& lo) {
    unsigned u  = __builtin_bit_cast(unsigned, x);
    unsigned short hs = (unsigned short)((u + 0x7fffu + ((u >> 16) & 1u)) >> 16);
    float hf = __builtin_bit_cast(float, (unsigned)hs << 16);
    float rf = x - hf;
    unsigned r  = __builtin_bit_cast(unsigned, rf);
    unsigned short ls = (unsigned short)((r + 0x7fffu + ((r >> 16) & 1u)) >> 16);
    hi = __builtin_bit_cast(__bf16, hs);
    lo = __builtin_bit_cast(__bf16, ls);
}

// K-permutation (involution) inside each 32-K group so that a lane's WMMA
// A-fragment (K {0..7,16..23} for lanes 0-15, {8..15,24..31} for lanes 16-31,
// per ISA 7.12.2) becomes 16 *contiguous* bf16 = one 32B load.
__device__ __forceinline__ int kperm(int i) {
    return (i < 8) ? i : (i < 16) ? i + 8 : (i < 24) ? i - 8 : i;
}

// ---------- operand-split kernels (run once per matrix) ----------

// A-operand: fp32 (rows x K, leading dim ld) -> swizzled hi/lo bf16 planes (ld=K)
__global__ __launch_bounds__(256) void split_a_kernel(
    const float* __restrict__ src, int ld, int kshift,
    __bf16* __restrict__ hi, __bf16* __restrict__ lo)
{
    int idx = blockIdx.x * 256 + threadIdx.x;
    int K = 1 << kshift;
    int r = idx >> kshift, k = idx & (K - 1);
    float v = src[(size_t)r * ld + k];
    __bf16 h, l; split_bf16(v, h, l);
    size_t o = ((size_t)r << kshift) + (k & ~31) + kperm(k & 31);
    hi[o] = h; lo[o] = l;
}

// B-operand: fp32 (rows x K) -> swizzled hi/lo bf16 planes (same kperm as A)
__global__ __launch_bounds__(256) void split_b_kernel(
    const float* __restrict__ src, __bf16* __restrict__ hi, __bf16* __restrict__ lo)
{
    int idx = blockIdx.x * 256 + threadIdx.x;
    __bf16 h, l; split_bf16(src[idx], h, l);
    int o = (idx & ~31) + kperm(idx & 31);
    hi[o] = h; lo[o] = l;
}

// ---------- GEMM: C[MxN] = A[MxK] * B[NxK]^T via bf16x3 WMMA ----------
// Operands are pre-split hi/lo bf16 planes (A swizzled). Each wave computes a
// 32x32 C quadrant = 2x2 WMMA tiles; 8 x 32B loads + 12 WMMAs per 32-K chunk.
// No LDS staging: on MI455X the whole working set is L2-resident (192 MB).

#define WMMA_BF16(AV, BV, CV) \
    __builtin_amdgcn_wmma_f32_16x16x32_bf16(false, (AV), false, (BV), (short)0, (CV), false, false)

enum { EPI_NONE = 0, EPI_SOFTPLUS_BIAS = 1 };

template <int EPI>
__global__ __launch_bounds__(128) void gemm_bf16x3_wmma(
    const __bf16* __restrict__ Ahi, const __bf16* __restrict__ Alo,
    const __bf16* __restrict__ Bhi, const __bf16* __restrict__ Blo,
    float* __restrict__ C, int ldc, int N, int K,
    const float* __restrict__ bias)
{
    const int lane = threadIdx.x & 31;
    const int wave = threadIdx.x >> 5;
    const int hsel = lane >> 4;           // lane half selects K sub-block
    const int lrow = lane & 15;

    const int mq = blockIdx.y * 64 + (wave >> 1) * 32;   // wave's 32-row quadrant
    const int nq = blockIdx.x * 64 + (wave & 1) * 32;    // wave's 32-col quadrant
    if (nq >= N) return;                  // wave-uniform (N is a multiple of 32)

    const size_t arow0 = (size_t)(mq + lrow) * K;        // A tile 0 (rows mq..mq+15)
    const size_t arow1 = arow0 + (size_t)16 * K;         // A tile 1
    const size_t brow0 = (size_t)(nq + lrow) * K;        // B tile 0
    const size_t brow1 = brow0 + (size_t)16 * K;         // B tile 1
    const int koff = hsel * 16;

    v8f c00 = {0,0,0,0,0,0,0,0}, c01 = c00, c10 = c00, c11 = c00;

    for (int k0 = 0; k0 < K; k0 += 32) {
        const int kb = k0 + koff;
        if (k0 + 32 < K) {                               // -> global_prefetch_b8
            __builtin_prefetch(Ahi + arow0 + kb + 32, 0, 3);
            __builtin_prefetch(Ahi + arow1 + kb + 32, 0, 3);
            __builtin_prefetch(Bhi + brow0 + kb + 32, 0, 3);
            __builtin_prefetch(Bhi + brow1 + kb + 32, 0, 3);
        }
        v16bf a0h = *(const v16bf*)(Ahi + arow0 + kb);
        v16bf a1h = *(const v16bf*)(Ahi + arow1 + kb);
        v16bf b0h = *(const v16bf*)(Bhi + brow0 + kb);
        v16bf b1h = *(const v16bf*)(Bhi + brow1 + kb);
        v16bf a0l = *(const v16bf*)(Alo + arow0 + kb);
        v16bf a1l = *(const v16bf*)(Alo + arow1 + kb);
        v16bf b0l = *(const v16bf*)(Blo + brow0 + kb);
        v16bf b1l = *(const v16bf*)(Blo + brow1 + kb);

        c00 = WMMA_BF16(a0h, b0h, c00); c01 = WMMA_BF16(a0h, b1h, c01);
        c10 = WMMA_BF16(a1h, b0h, c10); c11 = WMMA_BF16(a1h, b1h, c11);
        c00 = WMMA_BF16(a0h, b0l, c00); c01 = WMMA_BF16(a0h, b1l, c01);
        c10 = WMMA_BF16(a1h, b0l, c10); c11 = WMMA_BF16(a1h, b1l, c11);
        c00 = WMMA_BF16(a0l, b0h, c00); c01 = WMMA_BF16(a0l, b1h, c01);
        c10 = WMMA_BF16(a1l, b0h, c10); c11 = WMMA_BF16(a1l, b1h, c11);
    }

    v8f acc[2][2] = {{c00, c01}, {c10, c11}};
#pragma unroll
    for (int tj = 0; tj < 2; ++tj) {
        const int col = nq + tj * 16 + lrow;
        float bv = 0.f;
        if (EPI == EPI_SOFTPLUS_BIAS) bv = bias[col];
#pragma unroll
        for (int ti = 0; ti < 2; ++ti) {
#pragma unroll
            for (int r = 0; r < 8; ++r) {
                // C layout: VGPR r -> row r (lanes 0-15) / row r+8 (lanes 16-31)
                int row = mq + ti * 16 + r + hsel * 8;
                float v = acc[ti][tj][r];
                if (EPI == EPI_SOFTPLUS_BIAS) {
                    v += bv;
                    v = (v > 20.f) ? v : __logf(1.f + __expf(v));   // softplus
                }
                C[(size_t)row * ldc + col] = v;
            }
        }
    }
}

// ---------- depthwise causal conv1d(4) + bias + SiLU; emits fp32 + A-planes ----------
__global__ __launch_bounds__(256) void conv_silu_split_kernel(
    const float* __restrict__ xz, const float* __restrict__ cw,
    const float* __restrict__ cb, float* __restrict__ xa,
    __bf16* __restrict__ xahi, __bf16* __restrict__ xalo)
{
    int idx = blockIdx.x * 256 + threadIdx.x;      // over M_TOT * D_INNER
    int d = idx & (D_INNER - 1);
    int m = idx >> 11;                              // D_INNER == 2048
    int t = m & (SEQ - 1);
    float acc = cb[d];
#pragma unroll
    for (int j = 0; j < D_CONV; ++j) {
        int tt = t - (D_CONV - 1) + j;
        if (tt >= 0)
            acc += xz[(size_t)(m - (D_CONV - 1) + j) * (2 * D_INNER) + d] * cw[d * D_CONV + j];
    }
    float v = acc / (1.f + __expf(-acc));           // silu
    xa[idx] = v;
    __bf16 h, l; split_bf16(v, h, l);
    size_t o = (size_t)m * D_INNER + (d & ~31) + kperm(d & 31);
    xahi[o] = h; xalo[o] = l;
}

// ---------- selective scan ----------
// 16 lanes per channel (one per state n), 16 channels per 256-thread block.
#define SCAN_T 64
__global__ __launch_bounds__(256) void scan_kernel(
    const float* __restrict__ delta, const float* __restrict__ xa,
    const float* __restrict__ dbc, const float* __restrict__ A_log,
    float* __restrict__ y)
{
    __shared__ float sBC[SCAN_T][32];     // [t][0:16]=B, [t][16:32]=C
    __shared__ float sDelta[SCAN_T][16];
    __shared__ float sXa[SCAN_T][16];

    const int tid = threadIdx.x;
    const int n  = tid & 15;              // state index
    const int ch = tid >> 4;              // channel within block (0..15)
    const int b  = blockIdx.y;
    const int d  = blockIdx.x * 16 + ch;

    const float Aneg = -__expf(A_log[d * D_STATE + n]);   // A = -exp(A_log)
    float h = 0.f;

    for (int tc = 0; tc < SEQ; tc += SCAN_T) {
        for (int i = tid; i < SCAN_T * 32; i += 256) {
            int r = i >> 5, cix = i & 31;
            sBC[r][cix] = dbc[(size_t)(b * SEQ + tc + r) * DBC_COLS + DT_RANK + cix];
        }
        for (int i = tid; i < SCAN_T * 16; i += 256) {
            int r = i >> 4, cix = i & 15;
            size_t off = (size_t)(b * SEQ + tc + r) * D_INNER + blockIdx.x * 16 + cix;
            sDelta[r][cix] = delta[off];
            sXa[r][cix]    = xa[off];
        }
        __syncthreads();
        for (int s = 0; s < SCAN_T; ++s) {
            float dlt = sDelta[s][ch];
            float xv  = sXa[s][ch];
            float Bn  = sBC[s][n];
            float Cn  = sBC[s][16 + n];
            float dA  = __expf(dlt * Aneg);
            h = fmaf(dA, h, dlt * xv * Bn);           // h = dA*h + dBu
            float p = h * Cn;
            p += __shfl_xor(p, 1, 32);
            p += __shfl_xor(p, 2, 32);
            p += __shfl_xor(p, 4, 32);
            p += __shfl_xor(p, 8, 32);
            if (n == 0)
                y[(size_t)(b * SEQ + tc + s) * D_INNER + d] = p;
        }
        __syncthreads();
    }
}

// ---------- gating: yg = (y + D*xa) * silu(z); emits swizzled A-planes only ----------
__global__ __launch_bounds__(256) void gate_split_kernel(
    const float* __restrict__ y, const float* __restrict__ xa,
    const float* __restrict__ xz, const float* __restrict__ Dp,
    __bf16* __restrict__ yghi, __bf16* __restrict__ yglo)
{
    int idx = blockIdx.x * 256 + threadIdx.x;
    int d = idx & (D_INNER - 1);
    int m = idx >> 11;
    float z = xz[(size_t)m * (2 * D_INNER) + D_INNER + d];
    float g = z / (1.f + __expf(-z));
    float v = (y[idx] + Dp[d] * xa[idx]) * g;
    __bf16 h, l; split_bf16(v, h, l);
    size_t o = (size_t)m * D_INNER + (d & ~31) + kperm(d & 31);
    yghi[o] = h; yglo[o] = l;
}

extern "C" void kernel_launch(void* const* d_in, const int* in_sizes, int n_in,
                              void* d_out, int out_size, void* d_ws, size_t ws_size,
                              hipStream_t stream)
{
    const float* x        = (const float*)d_in[0];
    const float* in_proj  = (const float*)d_in[1];
    const float* conv_w   = (const float*)d_in[2];
    const float* conv_b   = (const float*)d_in[3];
    const float* x_proj   = (const float*)d_in[4];
    const float* dt_proj  = (const float*)d_in[5];
    const float* dt_b     = (const float*)d_in[6];
    const float* A_log    = (const float*)d_in[7];
    const float* Dp       = (const float*)d_in[8];
    const float* out_proj = (const float*)d_in[9];
    float* out = (float*)d_out;

    // Workspace carve-out (~280 MB). All sizes are multiples of 32B.
    char* p = (char*)d_ws;
    auto alloc_f = [&](size_t n) { float*  q = (float*)p;  p += n * 4; return q; };
    auto alloc_h = [&](size_t n) { __bf16* q = (__bf16*)p; p += n * 2; return q; };

    float* xz    = alloc_f((size_t)M_TOT * 2 * D_INNER);  // 4096 x 4096
    float* xa    = alloc_f((size_t)M_TOT * D_INNER);
    float* delta = alloc_f((size_t)M_TOT * D_INNER);
    float* yb    = alloc_f((size_t)M_TOT * D_INNER);
    float* dbc   = alloc_f((size_t)M_TOT * DBC_COLS);

    __bf16* xAh  = alloc_h((size_t)M_TOT * D_MODEL);      // x, swizzled A planes
    __bf16* xAl  = alloc_h((size_t)M_TOT * D_MODEL);
    __bf16* ipBh = alloc_h((size_t)2 * D_INNER * D_MODEL);// in_proj B planes
    __bf16* ipBl = alloc_h((size_t)2 * D_INNER * D_MODEL);
    __bf16* xaAh = alloc_h((size_t)M_TOT * D_INNER);      // xa A planes
    __bf16* xaAl = alloc_h((size_t)M_TOT * D_INNER);
    __bf16* xpBh = alloc_h((size_t)DBC_COLS * D_INNER);   // x_proj B planes
    __bf16* xpBl = alloc_h((size_t)DBC_COLS * D_INNER);
    __bf16* dlAh = alloc_h((size_t)M_TOT * DT_RANK);      // dl A planes
    __bf16* dlAl = alloc_h((size_t)M_TOT * DT_RANK);
    __bf16* dtBh = alloc_h((size_t)D_INNER * DT_RANK);    // dt_proj B planes
    __bf16* dtBl = alloc_h((size_t)D_INNER * DT_RANK);
    __bf16* ygAh = alloc_h((size_t)M_TOT * D_INNER);      // gated-y A planes
    __bf16* ygAl = alloc_h((size_t)M_TOT * D_INNER);
    __bf16* opBh = alloc_h((size_t)D_MODEL * D_INNER);    // out_proj B planes
    __bf16* opBl = alloc_h((size_t)D_MODEL * D_INNER);

    const dim3 gblk(128);   // 4 waves/block, one 32x32 C quadrant per wave

    // Operand splits for GEMM1
    split_a_kernel<<<(M_TOT * D_MODEL) / 256, 256, 0, stream>>>(x, D_MODEL, 10, xAh, xAl);
    split_b_kernel<<<(2 * D_INNER * D_MODEL) / 256, 256, 0, stream>>>(in_proj, ipBh, ipBl);

    // 1) xz = x @ in_proj_w.T       (M=4096, N=4096, K=1024)
    gemm_bf16x3_wmma<EPI_NONE><<<dim3((2 * D_INNER) / 64, M_TOT / 64), gblk, 0, stream>>>(
        xAh, xAl, ipBh, ipBl, xz, 2 * D_INNER, 2 * D_INNER, D_MODEL, nullptr);

    // 2) xa = silu(conv1d(xp) + conv_b); also emit xa A-planes
    conv_silu_split_kernel<<<(M_TOT * D_INNER) / 256, 256, 0, stream>>>(
        xz, conv_w, conv_b, xa, xaAh, xaAl);

    // 3) dbc = xa @ x_proj_w.T      (M=4096, N=96, K=2048)
    split_b_kernel<<<(DBC_COLS * D_INNER) / 256, 256, 0, stream>>>(x_proj, xpBh, xpBl);
    gemm_bf16x3_wmma<EPI_NONE><<<dim3(2, M_TOT / 64), gblk, 0, stream>>>(
        xaAh, xaAl, xpBh, xpBl, dbc, DBC_COLS, DBC_COLS, D_INNER, nullptr);

    // 4) delta = softplus(dl @ dt_proj_w.T + dt_proj_b)   (N=2048, K=64; dl = dbc[:, :64])
    split_a_kernel<<<(M_TOT * DT_RANK) / 256, 256, 0, stream>>>(dbc, DBC_COLS, 6, dlAh, dlAl);
    split_b_kernel<<<(D_INNER * DT_RANK) / 256, 256, 0, stream>>>(dt_proj, dtBh, dtBl);
    gemm_bf16x3_wmma<EPI_SOFTPLUS_BIAS><<<dim3(D_INNER / 64, M_TOT / 64), gblk, 0, stream>>>(
        dlAh, dlAl, dtBh, dtBl, delta, D_INNER, D_INNER, DT_RANK, dt_b);

    // 5) selective scan -> yb
    scan_kernel<<<dim3(D_INNER / 16, B_SZ), 256, 0, stream>>>(delta, xa, dbc, A_log, yb);

    // 6) yg = (yb + D*xa) * silu(z)  -> swizzled A-planes
    gate_split_kernel<<<(M_TOT * D_INNER) / 256, 256, 0, stream>>>(
        yb, xa, xz, Dp, ygAh, ygAl);

    // 7) out = yg @ out_proj_w.T    (M=4096, N=1024, K=2048)
    split_b_kernel<<<(D_MODEL * D_INNER) / 256, 256, 0, stream>>>(out_proj, opBh, opBl);
    gemm_bf16x3_wmma<EPI_NONE><<<dim3(D_MODEL / 64, M_TOT / 64), gblk, 0, stream>>>(
        ygAh, ygAl, opBh, opBl, out, D_MODEL, D_MODEL, D_INNER, nullptr);
}
